// GGNN_19619410608338
// MI455X (gfx1250) — matrix operand order repaired and checked by reading, not compiled, not run on hardware
//
#include <hip/hip_runtime.h>

typedef __attribute__((ext_vector_type(16))) _Float16 v16h;
typedef __attribute__((ext_vector_type(8)))  float    v8f;

#define B_   64
#define N_   512
#define H_   32
#define E_   4
#define M_   (N_*E_)     /* 2048 : inner (K) dimension of big GEMM */
#define MROW (2*M_)      /* 4096 : row stride of m                 */
#define NSTEPS 5

// ---------------------------------------------------------------------------
// Kernel A: s_in/s_out projections, reorder + transpose fused into the store.
// sT layout: [dir][b][hh][e*N + n]  (f16)  -> WMMA B-fragment columns are
// contiguous in K, so the GEMM kernel loads them as one 32-byte v16h.
// ---------------------------------------------------------------------------
__global__ void __launch_bounds__(256)
ggnn_proj(const float* __restrict__ h,
          const float* __restrict__ Win,  const float* __restrict__ bin,
          const float* __restrict__ Wout, const float* __restrict__ bout,
          _Float16* __restrict__ sT)
{
    __shared__ float hrow[H_];
    const int row = blockIdx.x;              // b*N + n
    const int b   = row >> 9;
    const int n   = row & (N_ - 1);
    const int t   = threadIdx.x;             // 0..255 : dir(1b) x k(7b)
    if (t < H_) hrow[t] = h[(size_t)row * H_ + t];
    __syncthreads();

    const int dir = t >> 7;
    const int k   = t & 127;
    const float* W  = dir ? Wout : Win;
    const float* bb = dir ? bout : bin;
    float acc = bb[k];
#pragma unroll
    for (int j = 0; j < H_; ++j) acc += W[k * H_ + j] * hrow[j];

    const int hh = k >> 2;                   // k = hh*E + e
    const int e  = k & 3;
    sT[((size_t)dir * B_ + b) * ((size_t)H_ * M_) + (size_t)hh * M_ + e * N_ + n] =
        (_Float16)acc;
}

// ---------------------------------------------------------------------------
// Kernel B: a[dir][b] = m_dir[b] (512x2048) @ s_reorder[b] (2048x32)
// One wave per 16x32 output tile; 64 K-steps of v_wmma_f32_16x16x32_f16 (x2).
// m is streamed from HBM exactly once per step (the bandwidth-dominant term),
// converted f32->f16 in-register.
// ---------------------------------------------------------------------------
__global__ void __launch_bounds__(256)
ggnn_gemm(const float* __restrict__ m,
          const _Float16* __restrict__ sT,
          float* __restrict__ aout)
{
    const int wave = (int)((blockIdx.x * blockDim.x + threadIdx.x) >> 5); // 0..4095
    const int lane = threadIdx.x & 31;
    const int half = lane >> 4;
    const int l16  = lane & 15;

    const int mt  = wave & 31;          // output row tile (16 rows)
    const int dir = (wave >> 5) & 1;    // in / out
    const int b   = wave >> 6;          // batch

    const float*    mrow = m  + (size_t)b * N_ * MROW
                              + (size_t)(mt * 16 + l16) * MROW + dir * M_;
    const _Float16* sb   = sT + ((size_t)dir * B_ + b) * ((size_t)H_ * M_);

    v8f c0 = {};
    v8f c1 = {};
    const int koffA = half * 16;  // A layout: lanes<16 -> K{0..15}, else K{16..31}

    for (int k0 = 0; k0 < M_; k0 += 32) {
        const float* ap = mrow + k0 + koffA;
        v16h af;
#pragma unroll
        for (int i = 0; i < 8; ++i) af[i]     = (_Float16)ap[i];
#pragma unroll
        for (int i = 0; i < 8; ++i) af[i + 8] = (_Float16)ap[8 + i];

        // B fragments: column = output channel, 16 contiguous K halves (32B)
        v16h b0 = *(const v16h*)(sb + (size_t)l16        * M_ + k0 + half * 16);
        v16h b1 = *(const v16h*)(sb + (size_t)(16 + l16) * M_ + k0 + half * 16);

        c0 = __builtin_amdgcn_wmma_f32_16x16x32_f16(false, af, false, b0,
                                                    (short)0, c0, false, false);
        c1 = __builtin_amdgcn_wmma_f32_16x16x32_f16(false, af, false, b1,
                                                    (short)0, c1, false, false);

        __builtin_prefetch(ap + 64, 0, 1);   // K tile two steps ahead (k0+64) of m -> global_prefetch_b8
    }

    float* dst = aout + ((size_t)dir * B_ + b) * ((size_t)N_ * H_)
                      + (size_t)mt * 16 * H_;
#pragma unroll
    for (int v = 0; v < 8; ++v) {            // D layout: VGPR v -> row v + half*8
        dst[(v + half * 8) * H_ + l16]       = c0[v];
        dst[(v + half * 8) * H_ + 16 + l16]  = c1[v];
    }
}

// ---------------------------------------------------------------------------
// Kernel C: GRU-style gate update, one wave (32 lanes = 32 channels) per row,
// 8 rows per 256-thread block. cat(96) staged in LDS.
// ---------------------------------------------------------------------------
__global__ void __launch_bounds__(256)
ggnn_gate(const float* __restrict__ ain, const float* __restrict__ aoutp,
          float* __restrict__ h,
          const float* __restrict__ Wz, const float* __restrict__ bz,
          const float* __restrict__ Wr, const float* __restrict__ br,
          const float* __restrict__ Wt, const float* __restrict__ bt)
{
    __shared__ float cat[8][96];
    const int sub  = threadIdx.x >> 5;
    const int lane = threadIdx.x & 31;
    const int row  = blockIdx.x * 8 + sub;

    const float av = ain  [(size_t)row * H_ + lane];
    const float ao = aoutp[(size_t)row * H_ + lane];
    const float hv = h    [(size_t)row * H_ + lane];
    cat[sub][lane]      = av;
    cat[sub][32 + lane] = ao;
    cat[sub][64 + lane] = hv;
    __syncthreads();

    float z = bz[lane];
    float r = br[lane];
#pragma unroll 4
    for (int j = 0; j < 96; ++j) {
        const float cj = cat[sub][j];
        z += cj * Wz[lane * 96 + j];
        r += cj * Wr[lane * 96 + j];
    }
    z = 1.0f / (1.0f + __expf(-z));
    r = 1.0f / (1.0f + __expf(-r));
    __syncthreads();
    cat[sub][64 + lane] = r * hv;
    __syncthreads();

    float tt = bt[lane];
#pragma unroll 4
    for (int j = 0; j < 96; ++j) tt += cat[sub][j] * Wt[lane * 96 + j];
    tt = tanhf(tt);

    h[(size_t)row * H_ + lane] = (1.0f - z) * hv + z * tt;
}

// ---------------------------------------------------------------------------
// Kernel D: output head. tanh([h,a] @ W1.T + b1) @ W2.T + b2 -> scalar/row.
// ---------------------------------------------------------------------------
__global__ void __launch_bounds__(256)
ggnn_head(const float* __restrict__ h, const float* __restrict__ a,
          const float* __restrict__ W1, const float* __restrict__ b1,
          const float* __restrict__ W2, const float* __restrict__ b2,
          float* __restrict__ out)
{
    __shared__ float hr[8][H_];
    const int sub  = threadIdx.x >> 5;
    const int lane = threadIdx.x & 31;
    const int row  = blockIdx.x * 8 + sub;

    hr[sub][lane] = h[(size_t)row * H_ + lane];
    __syncthreads();

    float acc = b1[lane];
#pragma unroll
    for (int j = 0; j < H_; ++j) acc += hr[sub][j] * W1[lane * 33 + j];
    acc += a[row] * W1[lane * 33 + 32];

    float part = tanhf(acc) * W2[lane];
#pragma unroll
    for (int off = 16; off > 0; off >>= 1) part += __shfl_xor(part, off, 32);
    if (lane == 0) out[row] = part + b2[0];
}

// ---------------------------------------------------------------------------
extern "C" void kernel_launch(void* const* d_in, const int* in_sizes, int n_in,
                              void* d_out, int out_size, void* d_ws, size_t ws_size,
                              hipStream_t stream)
{
    const float* x    = (const float*)d_in[0];
    const float* a    = (const float*)d_in[1];
    const float* m    = (const float*)d_in[2];
    const float* Win  = (const float*)d_in[3];
    const float* bin  = (const float*)d_in[4];
    const float* Wout = (const float*)d_in[5];
    const float* bout = (const float*)d_in[6];
    const float* Wz   = (const float*)d_in[7];
    const float* bz   = (const float*)d_in[8];
    const float* Wr   = (const float*)d_in[9];
    const float* br   = (const float*)d_in[10];
    const float* Wt   = (const float*)d_in[11];
    const float* bt   = (const float*)d_in[12];
    const float* W1   = (const float*)d_in[13];
    const float* b1   = (const float*)d_in[14];
    const float* W2   = (const float*)d_in[15];
    const float* b2   = (const float*)d_in[16];
    float* out = (float*)d_out;

    char* ws = (char*)d_ws;
    const size_t hBytes  = (size_t)B_ * N_ * H_ * sizeof(float);          // 4 MB
    const size_t sTBytes = (size_t)2 * B_ * H_ * M_ * sizeof(_Float16);   // 16 MB
    float*    h_buf = (float*)ws;
    _Float16* sT    = (_Float16*)(ws + hBytes);
    float*    a_buf = (float*)(ws + hBytes + sTBytes);                    // 8 MB
    float*    a_in  = a_buf;
    float*    a_out = a_buf + (size_t)B_ * N_ * H_;

    hipMemcpyAsync(h_buf, x, hBytes, hipMemcpyDeviceToDevice, stream);

    for (int step = 0; step < NSTEPS; ++step) {
        ggnn_proj<<<B_ * N_, 256, 0, stream>>>(h_buf, Win, bin, Wout, bout, sT);
        // 4096 waves (64 b x 2 dir x 32 row tiles) -> 512 blocks of 8 waves
        ggnn_gemm<<<512, 256, 0, stream>>>(m, sT, a_buf);
        ggnn_gate<<<(B_ * N_) / 8, 256, 0, stream>>>(a_in, a_out, h_buf,
                                                     Wz, bz, Wr, br, Wt, bt);
    }
    ggnn_head<<<(B_ * N_) / 8, 256, 0, stream>>>(h_buf, a, W1, b1, W2, b2, out);
}
